// SeparateWindowedValuesCoordinatesAttention_32581621908166
// MI455X (gfx1250) — matrix operand order repaired and checked by reading, not compiled, not run on hardware
//
#include <hip/hip_runtime.h>
#include <hip/hip_bf16.h>

typedef __attribute__((ext_vector_type(16))) __bf16 v16bf;
typedef __attribute__((ext_vector_type(8)))  float  v8f;

union FragAB {
  v16bf v;
  uint4 q[2];
};

#define WMMA_BF16(A, B, C) \
  __builtin_amdgcn_wmma_f32_16x16x32_bf16(false, (A), false, (B), (short)0, (C), false, false)

static __device__ __forceinline__ float red_sum16(float x) {
  x += __shfl_xor(x, 1); x += __shfl_xor(x, 2);
  x += __shfl_xor(x, 4); x += __shfl_xor(x, 8);
  return x;
}
static __device__ __forceinline__ float red_max16(float x) {
  x = fmaxf(x, __shfl_xor(x, 1)); x = fmaxf(x, __shfl_xor(x, 2));
  x = fmaxf(x, __shfl_xor(x, 4)); x = fmaxf(x, __shfl_xor(x, 8));
  return x;
}

// ---------------------------------------------------------------------------
// Kernel 0: weight prep.  wt_qkv[cc][k] = concat(w_qk,w_v)[k][cc] as bf16,
//           wt_out[vd][c] = w_out[c][vd] as bf16 (transposed => B-operand rows
//           are contiguous 16B chunks for WMMA fragment loads).
// ---------------------------------------------------------------------------
__global__ void prep_kernel(const float* __restrict__ w_qk,
                            const float* __restrict__ w_v,
                            const float* __restrict__ w_out,
                            __bf16* __restrict__ wt_qkv,
                            __bf16* __restrict__ wt_out)
{
  const int i = blockIdx.x * 256 + threadIdx.x;
  if (i < 768 * 256) {
    const int cc = i >> 8, kk = i & 255;
    const float x = (cc < 512) ? w_qk[kk * 512 + cc] : w_v[kk * 256 + (cc - 512)];
    wt_qkv[(long)cc * 256 + kk] = (__bf16)x;
  } else if (i < 768 * 256 + 256 * 256) {
    const int j = i - 768 * 256;
    const int vd = j >> 8, c = j & 255;
    wt_out[(long)vd * 256 + c] = (__bf16)w_out[c * 256 + vd];
  }
}

// ---------------------------------------------------------------------------
// Kernel 1: fused QKV projection GEMM + bias + RMS-norm(qk 512) + q/k/v
// interleave split (cat channel cc -> j=cc%3, c=cc/3, head=c%8, hd=c/8) +
// shift-roll(-4,-4) + window permutation on store.  V stored [hd][tok].
// Block: 32 tokens x 768 cols, 8 waves x (2x6) WMMA C-tiles, K = 8 x 32.
// ---------------------------------------------------------------------------
__global__ __launch_bounds__(256, 1)
void proj_kernel(const float* __restrict__ values,
                 const __bf16* __restrict__ wt,      // [768][256] bf16
                 const float* __restrict__ b_qk,
                 const float* __restrict__ rms_w,
                 const float* __restrict__ b_v,
                 __bf16* __restrict__ qws,
                 __bf16* __restrict__ kws,
                 __bf16* __restrict__ vws)
{
  __shared__ __align__(16) __bf16 As[32 * 256];
  __shared__ float rms_acc[32];

  const int tid  = threadIdx.x;
  const int wave = tid >> 5;
  const int lane = tid & 31;
  const int half = lane >> 4;
  const int l16  = lane & 15;
  const long tok0 = (long)blockIdx.x * 32;

  // Stage 32 tokens x 256 channels of `values` into LDS as bf16.
  {
    const float4* src = (const float4*)(values + tok0 * 256);
    #pragma unroll
    for (int i = 0; i < 8; ++i) {
      const int idx = tid + i * 256;             // float4 index (2048 total)
      const float4 f = src[idx];
      As[idx * 4 + 0] = (__bf16)f.x;
      As[idx * 4 + 1] = (__bf16)f.y;
      As[idx * 4 + 2] = (__bf16)f.z;
      As[idx * 4 + 3] = (__bf16)f.w;
    }
  }
  if (tid < 32) rms_acc[tid] = 0.0f;
  __syncthreads();

  const int colbase = wave * 96;
  v8f acc[2][6] = {};

  #pragma unroll 1
  for (int ks = 0; ks < 8; ++ks) {
    const int kb = ks * 32;
    FragAB a[2];
    #pragma unroll
    for (int mt = 0; mt < 2; ++mt) {
      // A element i <-> k = kb + 8*half + 16*(i>>3) + (i&7)
      const uint4* p = (const uint4*)&As[(mt * 16 + l16) * 256 + kb + 8 * half];
      a[mt].q[0] = p[0];
      a[mt].q[1] = p[2];
    }
    FragAB b[6];
    #pragma unroll
    for (int ct = 0; ct < 6; ++ct) {
      // B element i <-> k = kb + 8*half + 16*(i>>3) + (i&7)
      const uint4* p = (const uint4*)&wt[(long)(colbase + ct * 16 + l16) * 256 + kb + 8 * half];
      b[ct].q[0] = p[0];
      b[ct].q[1] = p[2];
    }
    if (ks + 1 < 8)
      __builtin_prefetch(&wt[(long)(colbase + l16) * 256 + kb + 32], 0, 1);
    #pragma unroll
    for (int mt = 0; mt < 2; ++mt)
      #pragma unroll
      for (int ct = 0; ct < 6; ++ct)
        acc[mt][ct] = WMMA_BF16(a[mt].v, b[ct].v, acc[mt][ct]);
  }

  // Bias before RMS (reference: qk = x@W + b, then normalize).
  #pragma unroll
  for (int ct = 0; ct < 6; ++ct) {
    const int cc = colbase + ct * 16 + l16;
    const float bias = (cc < 512) ? b_qk[cc] : b_v[cc - 512];
    #pragma unroll
    for (int mt = 0; mt < 2; ++mt)
      #pragma unroll
      for (int r = 0; r < 8; ++r)
        acc[mt][ct][r] += bias;
  }

  // Per-row sum of squares over the 512 qk columns.
  const int nqk = min(6, max(0, (512 - colbase) >> 4));
  if (nqk > 0) {
    #pragma unroll
    for (int mt = 0; mt < 2; ++mt) {
      #pragma unroll
      for (int r = 0; r < 8; ++r) {
        float ss = 0.0f;
        #pragma unroll
        for (int ct = 0; ct < 6; ++ct)
          if (ct < nqk) { const float x = acc[mt][ct][r]; ss += x * x; }
        ss = red_sum16(ss);
        if (l16 == 0) atomicAdd(&rms_acc[mt * 16 + r + 8 * half], ss);
      }
    }
  }
  __syncthreads();

  // Scatter-store q/k/v (bf16) with RMS scale, shift-roll and windowing.
  #pragma unroll
  for (int ct = 0; ct < 6; ++ct) {
    const int cc   = colbase + ct * 16 + l16;
    const int j    = cc % 3;
    const int c    = cc / 3;
    const int head = c & 7;
    const int hd   = c >> 3;
    const bool isqk = (cc < 512);
    const float rw = isqk ? rms_w[cc] : 0.0f;
    __bf16* dst = (j == 0) ? qws : ((j == 1) ? kws : vws);
    #pragma unroll
    for (int mt = 0; mt < 2; ++mt) {
      #pragma unroll
      for (int r = 0; r < 8; ++r) {
        const int m = mt * 16 + r + 8 * half;
        const long tok = tok0 + m;
        const int bb = (int)(tok >> 16);
        const int rr = ((int)(tok >> 8)) & 255;
        const int cl = (int)tok & 255;
        const int sr = (rr + 252) & 255;       // roll(-4)
        const int sc = (cl + 252) & 255;
        const int wx = sr >> 3, wy = sc >> 3;
        const int t  = (sr & 7) * 8 + (sc & 7);
        const long wbase = ((((long)bb * 32 + wx) * 32 + wy) * 8 + head) * 2048;
        float x = acc[mt][ct][r];
        if (isqk) {
          const float rs = rsqrtf(rms_acc[m] * (1.0f / 512.0f) + 1e-6f);
          x = x * rs * rw;
        }
        const long off = (j == 2) ? (wbase + (long)hd * 64 + t)   // V: [hd][tok]
                                  : (wbase + (long)t * 32 + hd);  // Q,K: [tok][hd]
        dst[off] = (__bf16)x;
      }
    }
  }
}

// ---------------------------------------------------------------------------
// Kernel 2: windowed attention.  One wave per (window, head).
// S = Q@K^T (4x4 tiles, K=HD=32 => single WMMA each), scale + rel-pos bias +
// shift masks + softmax in C-layout, P through per-wave LDS -> A layout,
// O = P@V (K=64 => two WMMA K-steps).  Output stored in shifted coords.
// ---------------------------------------------------------------------------
__global__ __launch_bounds__(256, 1)
void attn_kernel(const __bf16* __restrict__ qws,
                 const __bf16* __restrict__ kws,
                 const __bf16* __restrict__ vws,
                 const float* __restrict__ pos_emb,
                 __bf16* __restrict__ yws)
{
  __shared__ __align__(16) __bf16 Pl[8][64 * 64];   // 64 KiB

  const int tid  = threadIdx.x;
  const int wave = tid >> 5;
  const int lane = tid & 31;
  const int half = lane >> 4;
  const int l16  = lane & 15;

  const int blk = blockIdx.x;                 // bb*1024 + wx*32 + wy
  const int wy  = blk & 31;
  const int wx  = (blk >> 5) & 31;
  const int bb  = blk >> 10;
  const int head = wave;

  const long base = ((((long)bb * 32 + wx) * 32 + wy) * 8 + head) * 2048;
  const __bf16* q = qws + base;
  const __bf16* k = kws + base;
  const __bf16* v = vws + base;

  // ---- S = Q @ K^T ----
  v8f s[4][4] = {};
  {
    FragAB a[4], b[4];
    #pragma unroll
    for (int mt = 0; mt < 4; ++mt) {
      const uint4* p = (const uint4*)&q[(mt * 16 + l16) * 32 + 8 * half];
      a[mt].q[0] = p[0]; a[mt].q[1] = p[2];
    }
    #pragma unroll
    for (int nt = 0; nt < 4; ++nt) {
      const uint4* p = (const uint4*)&k[(nt * 16 + l16) * 32 + 8 * half];
      b[nt].q[0] = p[0]; b[nt].q[1] = p[2];
    }
    #pragma unroll
    for (int mt = 0; mt < 4; ++mt)
      #pragma unroll
      for (int nt = 0; nt < 4; ++nt)
        s[mt][nt] = WMMA_BF16(a[mt].v, b[nt].v, s[mt][nt]);
  }

  // ---- scale + bias + shift masks + softmax (row stats across 16 lanes) ----
  const float scale = 0.17677669529663687f;   // 1/sqrt(32)
  const bool rmask = (wx == 31);
  const bool cmask = (wy == 31);
  #pragma unroll
  for (int mt = 0; mt < 4; ++mt) {
    #pragma unroll
    for (int r = 0; r < 8; ++r) {
      const int ti = mt * 16 + r + 8 * half;
      float row[4];
      float mx = -3.0e38f;
      #pragma unroll
      for (int nt = 0; nt < 4; ++nt) {
        const int tj = nt * 16 + l16;
        const int r0 = ((tj >> 3) - (ti >> 3)) + 7;
        const int c0 = ((tj & 7) - (ti & 7)) + 7;
        float x = s[mt][nt][r] * scale + pos_emb[r0 * 15 + c0];
        if (rmask && (((ti ^ tj) & 32) != 0)) x += -1.0e9f;
        if (cmask && (((ti ^ tj) & 4)  != 0)) x += -1.0e9f;
        row[nt] = x;
        mx = fmaxf(mx, x);
      }
      mx = red_max16(mx);
      float sum = 0.0f;
      #pragma unroll
      for (int nt = 0; nt < 4; ++nt) {
        row[nt] = __expf(row[nt] - mx);
        sum += row[nt];
      }
      sum = red_sum16(sum);
      const float inv = 1.0f / sum;
      #pragma unroll
      for (int nt = 0; nt < 4; ++nt)
        Pl[wave][ti * 64 + nt * 16 + l16] = (__bf16)(row[nt] * inv);
    }
  }

  // ---- O = P @ V ----  (per-wave LDS region; DS ops are in-order per wave)
  v8f o[4][2] = {};
  #pragma unroll
  for (int kk = 0; kk < 2; ++kk) {
    FragAB pa[4], vb[2];
    #pragma unroll
    for (int mt = 0; mt < 4; ++mt) {
      const uint4* p = (const uint4*)&Pl[wave][(mt * 16 + l16) * 64 + kk * 32 + 8 * half];
      pa[mt].q[0] = p[0]; pa[mt].q[1] = p[2];
    }
    #pragma unroll
    for (int nt = 0; nt < 2; ++nt) {
      const uint4* p = (const uint4*)&v[(nt * 16 + l16) * 64 + kk * 32 + 8 * half];
      vb[nt].q[0] = p[0]; vb[nt].q[1] = p[2];
    }
    #pragma unroll
    for (int mt = 0; mt < 4; ++mt)
      #pragma unroll
      for (int nt = 0; nt < 2; ++nt)
        o[mt][nt] = WMMA_BF16(pa[mt].v, vb[nt].v, o[mt][nt]);
  }

  // ---- store y (shifted coords; reference never rolls back) ----
  #pragma unroll
  for (int mt = 0; mt < 4; ++mt)
    #pragma unroll
    for (int nt = 0; nt < 2; ++nt)
      #pragma unroll
      for (int r = 0; r < 8; ++r) {
        const int t  = mt * 16 + r + 8 * half;
        const int hd = nt * 16 + l16;
        const int sr = wx * 8 + (t >> 3);
        const int sc = wy * 8 + (t & 7);
        const long tok = ((long)bb << 16) + (sr << 8) + sc;
        yws[tok * 256 + head * 32 + hd] = (__bf16)o[mt][nt][r];
      }
}

// ---------------------------------------------------------------------------
// Kernel 3: output projection  out = y(bf16) @ w_out^T-layout(bf16) + b_out.
// Block: 32 tokens x 256 cols, 8 waves x (2x2) C-tiles, K = 8 x 32.
// ---------------------------------------------------------------------------
__global__ __launch_bounds__(256, 1)
void out_kernel(const __bf16* __restrict__ yws,
                const __bf16* __restrict__ wt,      // [256 vd][256 c] bf16
                const float* __restrict__ b_out,
                float* __restrict__ out)
{
  const int tid  = threadIdx.x;
  const int wave = tid >> 5;
  const int lane = tid & 31;
  const int half = lane >> 4;
  const int l16  = lane & 15;
  const long tok0 = (long)blockIdx.x * 32;
  const int colbase = wave * 32;

  v8f acc[2][2] = {};
  #pragma unroll 1
  for (int ks = 0; ks < 8; ++ks) {
    const int kb = ks * 32;
    FragAB a[2], b[2];
    #pragma unroll
    for (int mt = 0; mt < 2; ++mt) {
      const uint4* p = (const uint4*)&yws[(tok0 + mt * 16 + l16) * 256 + kb + 8 * half];
      a[mt].q[0] = p[0]; a[mt].q[1] = p[2];
    }
    #pragma unroll
    for (int ct = 0; ct < 2; ++ct) {
      const uint4* p = (const uint4*)&wt[(long)(colbase + ct * 16 + l16) * 256 + kb + 8 * half];
      b[ct].q[0] = p[0]; b[ct].q[1] = p[2];
    }
    if (ks + 1 < 8)
      __builtin_prefetch(&yws[(tok0 + l16) * 256 + kb + 32], 0, 1);
    #pragma unroll
    for (int mt = 0; mt < 2; ++mt)
      #pragma unroll
      for (int ct = 0; ct < 2; ++ct)
        acc[mt][ct] = WMMA_BF16(a[mt].v, b[ct].v, acc[mt][ct]);
  }

  #pragma unroll
  for (int ct = 0; ct < 2; ++ct) {
    const int vd = colbase + ct * 16 + l16;
    const float bo = b_out[vd];
    #pragma unroll
    for (int mt = 0; mt < 2; ++mt)
      #pragma unroll
      for (int r = 0; r < 8; ++r) {
        const long tok = tok0 + mt * 16 + r + 8 * half;
        out[tok * 256 + vd] = acc[mt][ct][r] + bo;
      }
  }
}

// ---------------------------------------------------------------------------
extern "C" void kernel_launch(void* const* d_in, const int* in_sizes, int n_in,
                              void* d_out, int out_size, void* d_ws, size_t ws_size,
                              hipStream_t stream)
{
  const float* values  = (const float*)d_in[0];
  const float* w_qk    = (const float*)d_in[1];
  const float* b_qk    = (const float*)d_in[2];
  const float* rms_w   = (const float*)d_in[3];
  const float* w_v     = (const float*)d_in[4];
  const float* b_v     = (const float*)d_in[5];
  const float* pos_emb = (const float*)d_in[6];
  const float* w_out   = (const float*)d_in[7];
  const float* b_out   = (const float*)d_in[8];
  float* out = (float*)d_out;

  char* ws = (char*)d_ws;
  const size_t OFF_WQKV = 0;
  const size_t OFF_WOUT = 768ull * 256 * 2;                  // 393216
  const size_t OFF_Q    = OFF_WOUT + 256ull * 256 * 2;       // 524288
  const size_t SZ_QKV   = 2ull * 32 * 32 * 8 * 64 * 32 * 2;  // 64 MiB each
  const size_t OFF_K    = OFF_Q + SZ_QKV;
  const size_t OFF_V    = OFF_K + SZ_QKV;
  const size_t OFF_Y    = OFF_V + SZ_QKV;                    // total ~257 MiB

  __bf16* wt_qkv = (__bf16*)(ws + OFF_WQKV);
  __bf16* wt_out = (__bf16*)(ws + OFF_WOUT);
  __bf16* qws = (__bf16*)(ws + OFF_Q);
  __bf16* kws = (__bf16*)(ws + OFF_K);
  __bf16* vws = (__bf16*)(ws + OFF_V);
  __bf16* yws = (__bf16*)(ws + OFF_Y);

  prep_kernel<<<1024, 256, 0, stream>>>(w_qk, w_v, w_out, wt_qkv, wt_out);
  proj_kernel<<<4096, 256, 0, stream>>>(values, wt_qkv, b_qk, rms_w, b_v, qws, kws, vws);
  attn_kernel<<<2048, 256, 0, stream>>>(qws, kws, vws, pos_emb, yws);
  out_kernel<<<4096, 256, 0, stream>>>(yws, wt_out, b_out, out);
}
